// LSTMLangModel_7464653160613
// MI455X (gfx1250) — matrix operand order repaired and checked by reading, not compiled, not run on hardware
//
#include <hip/hip_runtime.h>
#include <hip/hip_bf16.h>

// ---------------------------------------------------------------------------
// LSTM language model on gfx1250 (MI455X), bf16 WMMA path.
//   B=16, S=256, E=512, H=1024, V=32000
// Phases: prep/zero -> embed gather(bf16) -> weight transpose+bf16 ->
//         persistent 16-WG LSTM scan (grid barrier per step) ->
//         output projection GEMM (bf16 WMMA, f32 accum).
// ---------------------------------------------------------------------------

#define VOCAB  32000
#define EMBED  512
#define HIDDEN 1024
#define BATCH  16
#define SEQ    256
#define EH     (EMBED + HIDDEN)   // 1536
#define NGATE  (4 * HIDDEN)       // 4096 fused gate columns
#define NWG_LSTM 16               // persistent workgroups in the scan

typedef __attribute__((ext_vector_type(16))) __bf16 v16bf;
typedef __attribute__((ext_vector_type(8)))  __bf16 v8bf;
typedef __attribute__((ext_vector_type(8)))  float  v8f;

__device__ __forceinline__ v16bf cat8(v8bf lo, v8bf hi) {
  return __builtin_shufflevector(lo, hi, 0,1,2,3,4,5,6,7,8,9,10,11,12,13,14,15);
}

__device__ __forceinline__ float fast_sigmoid(float x) {
  return __builtin_amdgcn_rcpf(1.0f + __expf(-x));
}

__device__ __forceinline__ float fast_tanh(float x) {
#if __has_builtin(__builtin_amdgcn_tanhf)
  return __builtin_amdgcn_tanhf(x);          // native v_tanh_f32 on gfx1250
#else
  float e = __expf(2.0f * x);
  return 1.0f - 2.0f * __builtin_amdgcn_rcpf(e + 1.0f);
#endif
}

// ---------------------------------------------------------------------------
// Workspace zero-init (barrier counter + h double-buffer).
// ---------------------------------------------------------------------------
__global__ void lstm_prep_zero(unsigned* __restrict__ ctr,
                               __bf16*   __restrict__ hbuf) {
  if (blockIdx.x == 0 && threadIdx.x == 0) *ctr = 0u;
  unsigned* p = (unsigned*)hbuf;                 // 2*16*1024 bf16 = 16384 dwords
  const int n = 2 * BATCH * HIDDEN / 2;
  for (int i = threadIdx.x + blockIdx.x * 256; i < n; i += 256 * gridDim.x)
    p[i] = 0u;
}

// ---------------------------------------------------------------------------
// Embedding gather: tokens[b][s] -> X[s][b][e] in bf16 (A-matrix layout rows
// of the recurrent GEMM are batch indices, contiguous in e).
// ---------------------------------------------------------------------------
__global__ void embed_gather(const int*   __restrict__ tokens,
                             const float* __restrict__ emb,
                             __bf16*      __restrict__ X) {
  const int bs = blockIdx.x;            // b*SEQ + s
  const int b = bs >> 8, s = bs & 255;
  const int tok = tokens[bs];
  const float* src = emb + (size_t)tok * EMBED;
  __bf16* dst = X + (size_t)(s * BATCH + b) * EMBED;
  for (int e = threadIdx.x; e < EMBED; e += blockDim.x)
    dst[e] = (__bf16)src[e];
}

// ---------------------------------------------------------------------------
// Tiled transpose + f32->bf16: src[K][N] -> dst[N][K].  Grid (K/32, N/32),
// block (32,8).  Coalesced on both sides via LDS tile.
// ---------------------------------------------------------------------------
__global__ void transpose_to_bf16(const float* __restrict__ src,
                                  __bf16*      __restrict__ dst,
                                  int K, int N) {
  __shared__ float tile[32][33];
  const int k0 = blockIdx.x * 32, n0 = blockIdx.y * 32;
  #pragma unroll
  for (int r = 0; r < 4; ++r) {
    int k = k0 + threadIdx.y + r * 8;
    tile[threadIdx.y + r * 8][threadIdx.x] = src[(size_t)k * N + n0 + threadIdx.x];
  }
  __syncthreads();
  #pragma unroll
  for (int r = 0; r < 4; ++r) {
    int n = n0 + threadIdx.y + r * 8;
    dst[(size_t)n * K + k0 + threadIdx.x] =
        (__bf16)tile[threadIdx.x][threadIdx.y + r * 8];
  }
}

// ---------------------------------------------------------------------------
// Device-scope barrier across the 16 persistent LSTM workgroups.
// Monotonic counter (no reset needed within a launch).  The acq_rel atomic
// lowers to global_wb + s_wait_storecnt; the acquire spin re-invalidates the
// WGP cache so all waves of the block see fresh h after __syncthreads().
// ---------------------------------------------------------------------------
__device__ __forceinline__ void grid_barrier(unsigned* ctr, int step) {
  __syncthreads();
  if (threadIdx.x == 0) {
    __hip_atomic_fetch_add(ctr, 1u, __ATOMIC_ACQ_REL, __HIP_MEMORY_SCOPE_AGENT);
    const unsigned target = (unsigned)(NWG_LSTM * (step + 1));
    while (__hip_atomic_load(ctr, __ATOMIC_ACQUIRE, __HIP_MEMORY_SCOPE_AGENT) < target)
      __builtin_amdgcn_s_sleep(2);
  }
  __syncthreads();
}

// ---------------------------------------------------------------------------
// Persistent LSTM scan.  16 WGs x 256 threads (8 waves).  WG j owns h columns
// [j*64, j*64+64): 4 N-tiles per gate, 16 tiles total, 2 tiles per wave
// (both tiles of a wave belong to the same gate).  Per step:
//   gates[16 x 64*4] = [x_t | h_{t-1}] (16x1536 bf16) @ Wg^T slice  (WMMA bf16)
//   -> activations -> LDS -> elementwise c/h update -> h to global (bf16).
// ---------------------------------------------------------------------------
__global__ void lstm_scan(const __bf16* __restrict__ X,    // [SEQ][BATCH][EMBED]
                          const __bf16* __restrict__ Wgt,  // [NGATE][EH]
                          const float*  __restrict__ bfv,
                          const float*  __restrict__ biv,
                          const float*  __restrict__ bcv,
                          const float*  __restrict__ bov,
                          __bf16*       __restrict__ hbuf, // [2][BATCH][HIDDEN]
                          __bf16*       __restrict__ Hseq, // [BATCH][SEQ][HIDDEN]
                          unsigned*     __restrict__ ctr) {
  __shared__ float gl[4 * BATCH * 64];      // post-activation gates, 16 KB

  const int tid  = threadIdx.x;
  const int wgid = blockIdx.x;              // 0..15
  const int wave = tid >> 5;
  const int lane = tid & 31;
  const int lm   = lane & 15;               // A row (batch) / B column-in-tile
  const int half = lane >> 4;

  // two N-tiles per wave, same gate
  const int g   = wave >> 1;                // 0=f 1=i 2=c 3=o
  const int lc0 = (2 * wave) & 3;
  const int lc1 = (2 * wave + 1) & 3;
  const int n0  = g * HIDDEN + wgid * 64 + lc0 * 16 + lm;
  const int n1  = g * HIDDEN + wgid * 64 + lc1 * 16 + lm;
  const __bf16* Wrow0 = Wgt + (size_t)n0 * EH + half * 16;
  const __bf16* Wrow1 = Wgt + (size_t)n1 * EH + half * 16;

  const float* bptr = (g == 0) ? bfv : (g == 1) ? biv : (g == 2) ? bcv : bov;
  const float bias0 = bptr[wgid * 64 + lc0 * 16 + lm];
  const float bias1 = bptr[wgid * 64 + lc1 * 16 + lm];

  // elementwise ownership: thread -> (batch eb, 4 h-cols starting at ec)
  const int eb = tid >> 4;
  const int ec = (tid & 15) * 4;
  float cst[4] = {0.f, 0.f, 0.f, 0.f};      // persistent cell state

  const v8f vzero = {0.f, 0.f, 0.f, 0.f, 0.f, 0.f, 0.f, 0.f};

  for (int s = 0; s < SEQ; ++s) {
    const __bf16* hprev = hbuf + (size_t)(s & 1) * BATCH * HIDDEN
                               + (size_t)lm * HIDDEN;
    const __bf16* Xrow  = X + (size_t)(s * BATCH + lm) * EMBED;

    v8f acc0 = vzero, acc1 = vzero;

    // K tiles 0..15: x_t part (E = 512)
    #pragma unroll 4
    for (int kt = 0; kt < 16; ++kt) {
      const int k0 = kt * 32;
      v16bf a  = cat8(*(const v8bf*)(Xrow + k0 + half * 16),
                      *(const v8bf*)(Xrow + k0 + half * 16 + 8));
      v16bf b0 = cat8(*(const v8bf*)(Wrow0 + k0),
                      *(const v8bf*)(Wrow0 + k0 + 8));
      v16bf b1 = cat8(*(const v8bf*)(Wrow1 + k0),
                      *(const v8bf*)(Wrow1 + k0 + 8));
      acc0 = __builtin_amdgcn_wmma_f32_16x16x32_bf16(false, a, false, b0,
                                                     (short)0, acc0, false, false);
      acc1 = __builtin_amdgcn_wmma_f32_16x16x32_bf16(false, a, false, b1,
                                                     (short)0, acc1, false, false);
    }
    // K tiles 16..47: h_{t-1} part (H = 1024)
    #pragma unroll 4
    for (int kt = 16; kt < 48; ++kt) {
      const int k0 = kt * 32;
      const int kh = k0 - EMBED;
      v16bf a  = cat8(*(const v8bf*)(hprev + kh + half * 16),
                      *(const v8bf*)(hprev + kh + half * 16 + 8));
      v16bf b0 = cat8(*(const v8bf*)(Wrow0 + k0),
                      *(const v8bf*)(Wrow0 + k0 + 8));
      v16bf b1 = cat8(*(const v8bf*)(Wrow1 + k0),
                      *(const v8bf*)(Wrow1 + k0 + 8));
      acc0 = __builtin_amdgcn_wmma_f32_16x16x32_bf16(false, a, false, b0,
                                                     (short)0, acc0, false, false);
      acc1 = __builtin_amdgcn_wmma_f32_16x16x32_bf16(false, a, false, b1,
                                                     (short)0, acc1, false, false);
    }

    // bias + activation, stage into LDS per documented C-layout
    // (VGPR i: lanes 0-15 -> M=i, lanes 16-31 -> M=i+8; N = lane&15)
    #pragma unroll
    for (int i = 0; i < 8; ++i) {
      const int m = i + half * 8;             // batch row
      float r0 = acc0[i] + bias0;
      float r1 = acc1[i] + bias1;
      if (g == 2) { r0 = fast_tanh(r0);    r1 = fast_tanh(r1); }
      else        { r0 = fast_sigmoid(r0); r1 = fast_sigmoid(r1); }
      gl[g * 1024 + m * 64 + lc0 * 16 + lm] = r0;
      gl[g * 1024 + m * 64 + lc1 * 16 + lm] = r1;
    }
    __syncthreads();

    // elementwise LSTM update on this WG's 64-column slice
    const int hcol = wgid * 64 + ec;
    __bf16* hnew = hbuf + (size_t)((s + 1) & 1) * BATCH * HIDDEN
                        + (size_t)eb * HIDDEN + hcol;
    __bf16* hout = Hseq + ((size_t)eb * SEQ + s) * HIDDEN + hcol;
    #pragma unroll
    for (int j = 0; j < 4; ++j) {
      const int o = eb * 64 + ec + j;
      const float fv = gl[0 * 1024 + o];
      const float iv = gl[1 * 1024 + o];
      const float cd = gl[2 * 1024 + o];
      const float ov = gl[3 * 1024 + o];
      const float cn = cst[j] * fv + cd * iv;
      const float hv = cn * ov;               // reference: h = c * o (no tanh)
      cst[j] = cn;
      hnew[j] = (__bf16)hv;
      hout[j] = (__bf16)hv;
    }

    grid_barrier(ctr, s);    // makes h_t visible device-wide before step s+1
  }
}

// ---------------------------------------------------------------------------
// Output projection: out[r][v] = Hseq[r][:] @ W_out[:, v] + b_out[v]
//   r = b*SEQ + s (4096 rows), K = 1024, N = 32000.
// Wave computes a 64x64 block (4x4 WMMA tiles, 4x A and B fragment reuse);
// WG (8 waves, 2x4) covers 128x256; grid (32, 125).
// ---------------------------------------------------------------------------
__global__ void out_proj(const __bf16* __restrict__ Hseq, // [4096][1024]
                         const __bf16* __restrict__ Wt,   // [32000][1024]
                         const float*  __restrict__ bout,
                         float*        __restrict__ out) {
  const int tid  = threadIdx.x;
  const int wave = tid >> 5;
  const int lane = tid & 31;
  const int lm   = lane & 15;
  const int half = lane >> 4;
  const int wm   = wave >> 2;               // 0..1
  const int wn   = wave & 3;                // 0..3
  const int Mb   = blockIdx.x * 128 + wm * 64;
  const int Nb   = blockIdx.y * 256 + wn * 64;

  v8f acc[4][4] = {};
  const __bf16* arow[4];
  const __bf16* brow[4];
  #pragma unroll
  for (int mt = 0; mt < 4; ++mt)
    arow[mt] = Hseq + (size_t)(Mb + mt * 16 + lm) * HIDDEN + half * 16;
  #pragma unroll
  for (int nt = 0; nt < 4; ++nt)
    brow[nt] = Wt + (size_t)(Nb + nt * 16 + lm) * HIDDEN + half * 16;

  for (int kt = 0; kt < HIDDEN / 32; ++kt) {
    const int k0 = kt * 32;
    v16bf a[4], b[4];
    #pragma unroll
    for (int mt = 0; mt < 4; ++mt)
      a[mt] = cat8(*(const v8bf*)(arow[mt] + k0),
                   *(const v8bf*)(arow[mt] + k0 + 8));
    #pragma unroll
    for (int nt = 0; nt < 4; ++nt)
      b[nt] = cat8(*(const v8bf*)(brow[nt] + k0),
                   *(const v8bf*)(brow[nt] + k0 + 8));
    #pragma unroll
    for (int mt = 0; mt < 4; ++mt)
      #pragma unroll
      for (int nt = 0; nt < 4; ++nt)
        acc[mt][nt] = __builtin_amdgcn_wmma_f32_16x16x32_bf16(
            false, a[mt], false, b[nt], (short)0, acc[mt][nt], false, false);
  }

  // epilogue: bias add + f32 store
  float bias[4];
  #pragma unroll
  for (int nt = 0; nt < 4; ++nt) bias[nt] = bout[Nb + nt * 16 + lm];

  #pragma unroll
  for (int mt = 0; mt < 4; ++mt) {
    #pragma unroll
    for (int i = 0; i < 8; ++i) {
      const int r = Mb + mt * 16 + i + half * 8;
      float* orow = out + (size_t)r * VOCAB;
      #pragma unroll
      for (int nt = 0; nt < 4; ++nt)
        orow[Nb + nt * 16 + lm] = acc[mt][nt][i] + bias[nt];
    }
  }
}

// ---------------------------------------------------------------------------
// Host launcher.
// ---------------------------------------------------------------------------
extern "C" void kernel_launch(void* const* d_in, const int* in_sizes, int n_in,
                              void* d_out, int out_size, void* d_ws, size_t ws_size,
                              hipStream_t stream) {
  (void)in_sizes; (void)n_in; (void)out_size; (void)ws_size;

  const int*   tokens = (const int*)  d_in[0];
  const float* emb    = (const float*)d_in[1];
  const float* W_f    = (const float*)d_in[2];
  const float* b_f    = (const float*)d_in[3];
  const float* W_i    = (const float*)d_in[4];
  const float* b_i    = (const float*)d_in[5];
  const float* W_c    = (const float*)d_in[6];
  const float* b_c    = (const float*)d_in[7];
  const float* W_o    = (const float*)d_in[8];
  const float* b_o    = (const float*)d_in[9];
  const float* W_out  = (const float*)d_in[10];
  const float* b_out  = (const float*)d_in[11];
  float* out = (float*)d_out;

  // workspace carve-up (256B aligned): ~91 MB total
  char* ws = (char*)d_ws;
  size_t off = 0;
  auto carve = [&](size_t bytes) -> char* {
    char* p = ws + off;
    off = (off + bytes + 255) & ~(size_t)255;
    return p;
  };
  unsigned* ctr  = (unsigned*)carve(256);
  __bf16*   hbuf = (__bf16*)carve((size_t)2 * BATCH * HIDDEN * 2);
  __bf16*   X    = (__bf16*)carve((size_t)SEQ * BATCH * EMBED * 2);
  __bf16*   Hs   = (__bf16*)carve((size_t)BATCH * SEQ * HIDDEN * 2);
  __bf16*   Wgt  = (__bf16*)carve((size_t)NGATE * EH * 2);
  __bf16*   Wot  = (__bf16*)carve((size_t)VOCAB * HIDDEN * 2);

  lstm_prep_zero<<<dim3(8), dim3(256), 0, stream>>>(ctr, hbuf);
  embed_gather<<<dim3(BATCH * SEQ), dim3(256), 0, stream>>>(tokens, emb, X);

  const float* Wg[4] = {W_f, W_i, W_c, W_o};
  for (int g = 0; g < 4; ++g)
    transpose_to_bf16<<<dim3(EH / 32, HIDDEN / 32), dim3(32, 8), 0, stream>>>(
        Wg[g], Wgt + (size_t)g * HIDDEN * EH, EH, HIDDEN);
  transpose_to_bf16<<<dim3(HIDDEN / 32, VOCAB / 32), dim3(32, 8), 0, stream>>>(
      W_out, Wot, HIDDEN, VOCAB);

  lstm_scan<<<dim3(NWG_LSTM), dim3(256), 0, stream>>>(
      X, Wgt, b_f, b_i, b_c, b_o, hbuf, Hs, ctr);

  out_proj<<<dim3((BATCH * SEQ) / 128, VOCAB / 256), dim3(256), 0, stream>>>(
      Hs, Wot, b_out, out);
}
